// ColourVoxelGenerator_66623532695838
// MI455X (gfx1250) — matrix operand order repaired and checked by reading, not compiled, not run on hardware
//
#include <hip/hip_runtime.h>
#include <hip/hip_bf16.h>
#include <cmath>

// CDNA5 WMMA fragment types (wave32)
typedef __attribute__((ext_vector_type(16))) _Float16 v16h;
typedef __attribute__((ext_vector_type(8)))  _Float16 v8h;
typedef __attribute__((ext_vector_type(8)))  float    v8f;

#define DEV __device__ __forceinline__

DEV float lrelu_f(float x) { return x > 0.f ? x : 0.01f * x; }
DEV int imin_i(int a, int b) { return a < b ? a : b; }

// ---------------------------------------------------------------------------
// Linear: x = lrelu(encoded @ W + b), encoded (4,128), W (128,16384)
// ---------------------------------------------------------------------------
__global__ void k_linear(const float* __restrict__ e, const float* __restrict__ W,
                         const float* __restrict__ bias, float* __restrict__ out) {
  int idx = blockIdx.x * blockDim.x + threadIdx.x;
  if (idx >= 4 * 16384) return;
  int j = idx & 16383;
  int b = idx >> 14;
  float s = bias[j];
  const float* eb = e + b * 128;
  #pragma unroll 4
  for (int k = 0; k < 128; ++k) s = fmaf(eb[k], W[k * 16384 + j], s);
  out[idx] = lrelu_f(s);
}

// ---------------------------------------------------------------------------
// Activation layout change: f32 NCDHW -> f16 NDHWC (channels-last).
// Makes 8 consecutive channels a contiguous 16B run -> b128 A-fragment loads.
// ---------------------------------------------------------------------------
__global__ void k_tohalf_nhwc(const float* __restrict__ in, _Float16* __restrict__ oh,
                              int C, int S3) {
  long total = 4L * C * S3;
  long idx = (long)blockIdx.x * blockDim.x + threadIdx.x;
  if (idx >= total) return;
  int vox = (int)(idx % S3); long t = idx / S3;
  int c = (int)(t % C);
  int b = (int)(t / C);
  oh[((size_t)b * S3 + vox) * C + c] = (_Float16)in[idx];
}

// Weight repack: f32 OIDHW -> f16 [co][KPAD], k = tap*CIN + ci, zero padded.
__global__ void k_repackW(const float* __restrict__ w, _Float16* __restrict__ wh,
                          int CIN, int K3, int KPAD, long total) {
  long idx = (long)blockIdx.x * blockDim.x + threadIdx.x;
  if (idx >= total) return;
  int k  = (int)(idx % KPAD);
  int co = (int)(idx / KPAD);
  float v = 0.f;
  if (k < CIN * K3) {
    int ci = k % CIN, tap = k / CIN;
    v = w[((size_t)co * CIN + ci) * K3 + tap];
  }
  wh[idx] = (_Float16)v;
}

// ---------------------------------------------------------------------------
// Implicit-GEMM 3D conv via V_WMMA_F32_16X16X32_F16, N-register-blocked.
// inH: (B, S^3, CIN) f16 channels-last; wh: (COUT, KPAD) f16 packed.
// Tap-outer loop: CIN % 32 == 0 so every 32-chunk is inside one tap; the
// boundary/reflect logic runs once per tap (27x total), tap decode is
// incremental (no div/mod), and the K loop is completely guard-free.
// A half-fragment = one b128 load; B fragment = two b128 loads per N tile.
// ---------------------------------------------------------------------------
template<int CIN, int COUT, int S, int KS, int PAD, bool REFLECT, int NB>
__global__ __launch_bounds__(32)
void k_conv_wmma(const _Float16* __restrict__ inH, const _Float16* __restrict__ wh,
                 const float* __restrict__ bias, float* __restrict__ out) {
  static_assert(COUT % (16 * NB) == 0, "N blocking must divide COUT");
  static_assert(CIN % 32 == 0, "chunk must sit inside one tap");
  constexpr int S3   = S * S * S;
  constexpr int K3   = KS * KS * KS;
  constexpr int KPAD = CIN * K3;            // multiple of 32

  const int lane  = threadIdx.x;
  const int mloc  = lane & 15;
  const int hi    = lane >> 4;
  const int mtile = blockIdx.x;
  const int b     = blockIdx.z;

  const int vox = mtile * 16 + mloc;
  const int x = vox % S;
  const int y = (vox / S) % S;
  const int z = vox / (S * S);

  const int co0 = blockIdx.y * (16 * NB) + mloc;

  const _Float16* __restrict__ inB = inH + (size_t)b * S3 * CIN;

  v8f acc[NB];
  #pragma unroll
  for (int nb = 0; nb < NB; ++nb) acc[nb] = (v8f){};

  int dzt = 0, dyt = 0, dxt = 0;            // incremental tap decode
  for (int tap = 0; tap < K3; ++tap) {
    int zz = z + dzt - PAD, yy = y + dyt - PAD, xx = x + dxt - PAD;
    bool ok = true;
    if (REFLECT) {
      zz = zz < 0 ? -zz : (zz >= S ? 2 * S - 2 - zz : zz);
      yy = yy < 0 ? -yy : (yy >= S ? 2 * S - 2 - yy : yy);
      xx = xx < 0 ? -xx : (xx >= S ? 2 * S - 2 - xx : xx);
    } else {
      ok = (unsigned)zz < (unsigned)S && (unsigned)yy < (unsigned)S &&
           (unsigned)xx < (unsigned)S;
    }
    const _Float16* ap = inB + (size_t)((zz * S + yy) * S + xx) * CIN;

    for (int c0 = 0; c0 < CIN; c0 += 32) {
      v16h av;
      if (ok) {
        v8h a0 = *(const v8h*)(ap + c0 + hi * 8);        // global_load_b128
        v8h a1 = *(const v8h*)(ap + c0 + 16 + hi * 8);   // global_load_b128
        #pragma unroll
        for (int l = 0; l < 8; ++l) { av[l] = a0[l]; av[l + 8] = a1[l]; }
      } else {
        #pragma unroll
        for (int l = 0; l < 16; ++l) av[l] = (_Float16)0.f;
      }
      const int k0 = tap * CIN + c0;
      #pragma unroll
      for (int nb = 0; nb < NB; ++nb) {
        const _Float16* wp = wh + (size_t)(co0 + nb * 16) * KPAD + k0 + hi * 8;
        v8h b0 = *(const v8h*)(wp);                      // global_load_b128
        v8h b1 = *(const v8h*)(wp + 16);                 // global_load_b128
        v16h bv;
        #pragma unroll
        for (int l = 0; l < 8; ++l) { bv[l] = b0[l]; bv[l + 8] = b1[l]; }
        acc[nb] = __builtin_amdgcn_wmma_f32_16x16x32_f16(false, av, false, bv,
                                                         (short)0, acc[nb],
                                                         false, false);
      }
    }
    if (++dxt == KS) { dxt = 0; if (++dyt == KS) { dyt = 0; ++dzt; } }
  }

  #pragma unroll
  for (int nb = 0; nb < NB; ++nb) {
    const int co = co0 + nb * 16;
    const float bval = bias[co];
    #pragma unroll
    for (int r = 0; r < 8; ++r) {
      const int mm = mtile * 16 + r + 8 * hi;
      out[((size_t)(b * COUT + co)) * S3 + mm] = acc[nb][r] + bval;
    }
  }
}

// ---------------------------------------------------------------------------
// Residual 5x5x5 conv (CIN=COUT=16, S=64, zero pad 2), LDS-staged.
// 4 waves per WG own one x-row of 64 outputs at fixed (b,z,y). Staged halo is
// 26(plane) x 68(x) x 16(ci) f16 (55.3 KB of the 320 KB WGP LDS); plane 25 is
// a zero plane so the padded K tail (taps >= 125) needs no guard. Chunk = 2
// taps (CIN=16); tap decode is carry-incremental. Inner loop: 2 ds_load_b128
// (A) + 2 global_load_b128 (packed f16 B) + 1 WMMA, guard-free.
// ---------------------------------------------------------------------------
__global__ __launch_bounds__(128)
void k_conv5_res(const _Float16* __restrict__ inH,   // (B, S^3, 16) f16
                 const _Float16* __restrict__ wh,    // (16, 2016) f16 packed
                 const float* __restrict__ bias, float* __restrict__ out) {
  constexpr int S = 64, S3 = S * S * S, CIN = 16;
  constexpr int KPAD = 2016, XH = 68, NPL = 26;
  __shared__ _Float16 sh[NPL * XH * CIN];

  const int b  = blockIdx.z;
  const int zy = blockIdx.x;                // z*64 + y
  const int y  = zy & 63;
  const int z  = zy >> 6;

  const _Float16* __restrict__ inB = inH + (size_t)b * S3 * CIN;

  // Vectorized zero-padded staging: one (plane, x) = 16 f16 = 2x b128.
  for (int t = threadIdx.x; t < NPL * XH; t += 128) {
    int xx = t % XH;
    int pl = t / XH;                        // 25 = zero plane
    int dy = pl % 5, dz = pl / 5;
    int zi = z + dz - 2, yi = y + dy - 2, xi = xx - 2;
    v8h a0 = {}, a1 = {};
    if (pl < 25 && (unsigned)zi < (unsigned)S && (unsigned)yi < (unsigned)S &&
        (unsigned)xi < (unsigned)S) {
      const _Float16* src = inB + (size_t)((zi * S + yi) * S + xi) * CIN;
      a0 = *(const v8h*)(src);
      a1 = *(const v8h*)(src + 8);
    }
    *(v8h*)(sh + t * 16)     = a0;          // ds_store_b128
    *(v8h*)(sh + t * 16 + 8) = a1;
  }
  __syncthreads();

  const int wave = threadIdx.x >> 5;
  const int lane = threadIdx.x & 31;
  const int mloc = lane & 15;
  const int hi   = lane >> 4;
  const int xb   = wave * 16 + mloc;

  const _Float16* __restrict__ wp0 = wh + (size_t)mloc * KPAD;  // co = mloc

  v8f acc = {};
  int tz = 0, ty = 0, tx = 0;               // decode of tap t = 2*chunk
  for (int k0 = 0; k0 < KPAD; k0 += 32) {
    int tx1 = tx + 1, ty1 = ty, tz1 = tz;   // tap t+1 (may be zero plane 25)
    if (tx1 == 5) { tx1 = 0; if (++ty1 == 5) { ty1 = 0; ++tz1; } }
    const int s0 = ((tz * 5 + ty)  * XH + (xb + tx))  * CIN + hi * 8;
    const int s1 = ((tz1 * 5 + ty1) * XH + (xb + tx1)) * CIN + hi * 8;
    v8h a0 = *(const v8h*)(sh + s0);        // ds_load_b128
    v8h a1 = *(const v8h*)(sh + s1);        // ds_load_b128
    v16h av;
    #pragma unroll
    for (int l = 0; l < 8; ++l) { av[l] = a0[l]; av[l + 8] = a1[l]; }
    const _Float16* wp = wp0 + k0 + hi * 8;
    v8h b0 = *(const v8h*)(wp);             // global_load_b128
    v8h b1 = *(const v8h*)(wp + 16);
    v16h bv;
    #pragma unroll
    for (int l = 0; l < 8; ++l) { bv[l] = b0[l]; bv[l + 8] = b1[l]; }
    acc = __builtin_amdgcn_wmma_f32_16x16x32_f16(false, av, false, bv,
                                                 (short)0, acc, false, false);
    tx += 2;
    if (tx >= 5) { tx -= 5; if (++ty == 5) { ty = 0; ++tz; } }
  }

  const float bval = bias[mloc];
  #pragma unroll
  for (int r = 0; r < 8; ++r) {
    const int xo = wave * 16 + r + 8 * hi;
    out[((size_t)(b * 16 + mloc)) * S3 + (z * S + y) * S + xo] = acc[r] + bval;
  }
}

// ---------------------------------------------------------------------------
// BatchNorm statistics: per-channel mean / biased variance over (B, S^3)
// ---------------------------------------------------------------------------
__global__ void k_bnstats(const float* __restrict__ x, float* __restrict__ st,
                          int C, int S3) {
  __shared__ float sh[256];
  __shared__ float sh2[256];
  const int c = blockIdx.x;
  float s = 0.f, s2 = 0.f;
  for (int b = 0; b < 4; ++b) {
    const float* p = x + ((size_t)b * C + c) * S3;
    for (int v = threadIdx.x; v < S3; v += blockDim.x) {
      float t = p[v]; s += t; s2 += t * t;
    }
  }
  sh[threadIdx.x] = s; sh2[threadIdx.x] = s2;
  __syncthreads();
  for (int off = 128; off > 0; off >>= 1) {
    if ((int)threadIdx.x < off) {
      sh[threadIdx.x]  += sh[threadIdx.x + off];
      sh2[threadIdx.x] += sh2[threadIdx.x + off];
    }
    __syncthreads();
  }
  if (threadIdx.x == 0) {
    float n = 4.0f * (float)S3;
    float m = sh[0] / n;
    st[2 * c]     = m;
    st[2 * c + 1] = sh2[0] / n - m * m;
  }
}

__global__ void k_bnapply(const float* __restrict__ in, float* __restrict__ out,
                          const float* __restrict__ st, const float* __restrict__ g,
                          const float* __restrict__ be, int C, int S3, long total,
                          int do_lrelu) {
  long idx = (long)blockIdx.x * blockDim.x + threadIdx.x;
  if (idx >= total) return;
  int c = (int)((idx / S3) % C);
  float m = st[2 * c], v = st[2 * c + 1];
  float y = (in[idx] - m) * rsqrtf(v + 1e-5f) * g[c] + be[c];
  if (do_lrelu) y = lrelu_f(y);
  out[idx] = y;
}

// ---------------------------------------------------------------------------
// Trilinear 2x upsample, half-pixel convention (jax.image.resize 'trilinear')
// ---------------------------------------------------------------------------
__global__ void k_upsample2x(const float* __restrict__ in, float* __restrict__ out,
                             int C, int S) {
  const int S2 = S * 2;
  const long total = 4L * C * S2 * S2 * S2;
  long idx = (long)blockIdx.x * blockDim.x + threadIdx.x;
  if (idx >= total) return;
  int xo = (int)(idx % S2); long t = idx / S2;
  int yo = (int)(t % S2);   t /= S2;
  int zo = (int)(t % S2);   t /= S2;
  int c  = (int)(t % C);
  int b  = (int)(t / C);
  const float* p = in + ((size_t)b * C + c) * S * S * S;
  float fz = fminf(fmaxf((zo + 0.5f) * 0.5f - 0.5f, 0.f), (float)(S - 1));
  float fy = fminf(fmaxf((yo + 0.5f) * 0.5f - 0.5f, 0.f), (float)(S - 1));
  float fx = fminf(fmaxf((xo + 0.5f) * 0.5f - 0.5f, 0.f), (float)(S - 1));
  int z0 = (int)fz, y0 = (int)fy, x0 = (int)fx;
  float gz = fz - z0, gy = fy - y0, gx = fx - x0;
  int z1 = imin_i(z0 + 1, S - 1), y1 = imin_i(y0 + 1, S - 1), x1 = imin_i(x0 + 1, S - 1);
  auto at = [&](int zz, int yy, int xx) { return p[(zz * S + yy) * S + xx]; };
  float v00 = at(z0, y0, x0) + gx * (at(z0, y0, x1) - at(z0, y0, x0));
  float v01 = at(z0, y1, x0) + gx * (at(z0, y1, x1) - at(z0, y1, x0));
  float v10 = at(z1, y0, x0) + gx * (at(z1, y0, x1) - at(z1, y0, x0));
  float v11 = at(z1, y1, x0) + gx * (at(z1, y1, x1) - at(z1, y1, x0));
  float v0 = v00 + gy * (v01 - v00);
  float v1 = v10 + gy * (v11 - v10);
  out[idx] = v0 + gz * (v1 - v0);
}

__global__ void k_addlrelu(float* __restrict__ x, const float* __restrict__ h,
                           long total) {
  long idx = (long)blockIdx.x * blockDim.x + threadIdx.x;
  if (idx >= total) return;
  x[idx] = lrelu_f(x[idx] + h[idx]);
}

// 1x1x1 conv 16->4 + sigmoid
__global__ void k_outconv_sig(const float* __restrict__ x, const float* __restrict__ w,
                              const float* __restrict__ ob, float* __restrict__ vox) {
  const int S3 = 64 * 64 * 64;
  const long total = 4L * 4 * S3;
  long idx = (long)blockIdx.x * blockDim.x + threadIdx.x;
  if (idx >= total) return;
  int v  = (int)(idx % S3); long t = idx / S3;
  int co = (int)(t % 4);
  int b  = (int)(t / 4);
  float s = ob[co];
  #pragma unroll
  for (int ci = 0; ci < 16; ++ci)
    s = fmaf(x[((size_t)(b * 16 + ci)) * S3 + v], w[co * 16 + ci], s);
  vox[idx] = 1.f / (1.f + expf(-s));
}

// ---------------------------------------------------------------------------
// Rotation resample: src = R @ (g - ctr) + ctr, trilinear, constant 0 outside
// ---------------------------------------------------------------------------
__global__ void k_rotate(const float* __restrict__ vox, const float* __restrict__ ang,
                         float* __restrict__ rot) {
  const int S = 64; const int S3 = S * S * S;
  const long total = 4L * S3;
  long idx = (long)blockIdx.x * blockDim.x + threadIdx.x;
  if (idx >= total) return;
  int k  = (int)(idx % S); long t = idx / S;
  int j  = (int)(t % S);   t /= S;
  int iq = (int)(t % S);
  int b  = (int)(t / S);
  float cx = cosf(ang[b * 3 + 0]), sx = sinf(ang[b * 3 + 0]);
  float cy = cosf(ang[b * 3 + 1]), sy = sinf(ang[b * 3 + 1]);
  float cz = cosf(ang[b * 3 + 2]), sz = sinf(ang[b * 3 + 2]);
  float R00 = cz * cy, R01 = cz * sy * sx - sz * cx, R02 = cz * sy * cx + sz * sx;
  float R10 = sz * cy, R11 = sz * sy * sx + cz * cx, R12 = sz * sy * cx - cz * sx;
  float R20 = -sy,     R21 = cy * sx,                R22 = cy * cx;
  const float ctr = (S - 1) * 0.5f;
  float p0 = iq - ctr, p1 = j - ctr, p2 = k - ctr;
  float c0 = R00 * p0 + R01 * p1 + R02 * p2 + ctr;
  float c1 = R10 * p0 + R11 * p1 + R12 * p2 + ctr;
  float c2 = R20 * p0 + R21 * p1 + R22 * p2 + ctr;
  int i0 = (int)floorf(c0), i1 = (int)floorf(c1), i2 = (int)floorf(c2);
  float f0 = c0 - i0, f1 = c1 - i1, f2 = c2 - i2;
  float a4[4] = {0.f, 0.f, 0.f, 0.f};
  const float* vb = vox + (size_t)b * 4 * S3;
  #pragma unroll
  for (int d0 = 0; d0 < 2; ++d0)
    #pragma unroll
    for (int d1 = 0; d1 < 2; ++d1)
      #pragma unroll
      for (int d2 = 0; d2 < 2; ++d2) {
        int a0 = i0 + d0, a1 = i1 + d1, a2 = i2 + d2;
        float w = (d0 ? f0 : 1.f - f0) * (d1 ? f1 : 1.f - f1) * (d2 ? f2 : 1.f - f2);
        if ((unsigned)a0 < (unsigned)S && (unsigned)a1 < (unsigned)S &&
            (unsigned)a2 < (unsigned)S) {
          size_t o = (size_t)(a0 * S + a1) * S + a2;
          a4[0] += w * vb[0 * (size_t)S3 + o];
          a4[1] += w * vb[1 * (size_t)S3 + o];
          a4[2] += w * vb[2 * (size_t)S3 + o];
          a4[3] += w * vb[3 * (size_t)S3 + o];
        }
      }
  size_t po = (size_t)(iq * S + j) * S + k;
  rot[(size_t)b * 4 * S3 + 0 * (size_t)S3 + po] = a4[0];
  rot[(size_t)b * 4 * S3 + 1 * (size_t)S3 + po] = a4[1];
  rot[(size_t)b * 4 * S3 + 2 * (size_t)S3 + po] = a4[2];
  rot[(size_t)b * 4 * S3 + 3 * (size_t)S3 + po] = a4[3];
}

// Alpha compositing along z (reversed), out = 2*img - 1
__global__ void k_composite(const float* __restrict__ rot, float* __restrict__ out) {
  const int S = 64; const int S3 = S * S * S;
  int idx = blockIdx.x * blockDim.x + threadIdx.x;
  if (idx >= 4 * S * S) return;
  int j  = idx % S; int t = idx / S;
  int iq = t % S;   int b = t / S;
  const float* base = rot + (size_t)b * 4 * S3;
  size_t pix = (size_t)(iq * S + j) * S;
  float r = 0.f, g = 0.f, bl = 0.f;
  for (int kk = 0; kk < S; ++kk) {
    int z = S - 1 - kk;
    float a  = base[3 * (size_t)S3 + pix + z];
    float om = 1.f - a;
    r  = r  * om + a * base[0 * (size_t)S3 + pix + z];
    g  = g  * om + a * base[1 * (size_t)S3 + pix + z];
    bl = bl * om + a * base[2 * (size_t)S3 + pix + z];
  }
  out[((size_t)(b * 3 + 0) * S + iq) * S + j] = 2.f * r  - 1.f;
  out[((size_t)(b * 3 + 1) * S + iq) * S + j] = 2.f * g  - 1.f;
  out[((size_t)(b * 3 + 2) * S + iq) * S + j] = 2.f * bl - 1.f;
}

// ---------------------------------------------------------------------------
extern "C" void kernel_launch(void* const* d_in, const int* in_sizes, int n_in,
                              void* d_out, int out_size, void* d_ws, size_t ws_size,
                              hipStream_t stream) {
  (void)in_sizes; (void)n_in; (void)out_size; (void)ws_size;

  const float* encoded = (const float*)d_in[0];
  const float* angles  = (const float*)d_in[1];
  int i = 2;
  struct Blk { const float *b1, *b2, *be1, *be2, *g1, *g2, *w1, *w2; } blk[4];
  for (int t = 0; t < 4; ++t) {
    blk[t].b1  = (const float*)d_in[i++]; blk[t].b2  = (const float*)d_in[i++];
    blk[t].be1 = (const float*)d_in[i++]; blk[t].be2 = (const float*)d_in[i++];
    blk[t].g1  = (const float*)d_in[i++]; blk[t].g2  = (const float*)d_in[i++];
    blk[t].w1  = (const float*)d_in[i++]; blk[t].w2  = (const float*)d_in[i++];
  }
  const float* lin_b = (const float*)d_in[i++];
  const float* lin_w = (const float*)d_in[i++];
  const float* out_b = (const float*)d_in[i++];
  const float* out_w = (const float*)d_in[i++];
  const float *rba  = (const float*)d_in[i++], *rbb  = (const float*)d_in[i++];
  const float *rbea = (const float*)d_in[i++], *rbeb = (const float*)d_in[i++];
  const float *rga  = (const float*)d_in[i++], *rgb  = (const float*)d_in[i++];
  const float *rwa  = (const float*)d_in[i++], *rwb  = (const float*)d_in[i++];

  float* B0 = (float*)d_ws;               // 33,554,432 f32
  float* B1 = B0 + 33554432;              // 33,554,432 f32
  float* MD = B1 + 33554432;              // 16,777,216 f32
  float* ST = MD + 16777216;              // 512 f32 (mean/var pairs)
  _Float16* AH = (_Float16*)(ST + 512);   // 33,554,432 f16 (channels-last act)
  _Float16* WH = AH + 33554432;           // 2,097,152 f16 (packed weights)
  float* out = (float*)d_out;

  auto gr = [](long total) { return (unsigned)((total + 255) / 256); };

  k_linear<<<gr(65536), 256, 0, stream>>>(encoded, lin_w, lin_b, B0);

  // ---- Block 1: C=256, S=4 -> 128@8^3 ----
  k_tohalf_nhwc<<<gr(4L * 256 * 64), 256, 0, stream>>>(B0, AH, 256, 64);
  k_repackW<<<gr(256L * 6912), 256, 0, stream>>>(blk[0].w1, WH, 256, 27, 6912, 256L * 6912);
  k_conv_wmma<256, 256, 4, 3, 1, false, 4><<<dim3(4, 4, 4), 32, 0, stream>>>(AH, WH, blk[0].b1, B1);
  k_bnstats<<<256, 256, 0, stream>>>(B1, ST, 256, 64);
  k_bnapply<<<gr(4L * 256 * 64), 256, 0, stream>>>(B1, B1, ST, blk[0].g1, blk[0].be1, 256, 64, 4L * 256 * 64, 1);
  k_upsample2x<<<gr(4L * 256 * 512), 256, 0, stream>>>(B1, B0, 256, 4);
  k_tohalf_nhwc<<<gr(4L * 256 * 512), 256, 0, stream>>>(B0, AH, 256, 512);
  k_repackW<<<gr(128L * 6912), 256, 0, stream>>>(blk[0].w2, WH, 256, 27, 6912, 128L * 6912);
  k_conv_wmma<256, 128, 8, 3, 1, true, 4><<<dim3(32, 2, 4), 32, 0, stream>>>(AH, WH, blk[0].b2, B1);
  k_bnstats<<<128, 256, 0, stream>>>(B1, ST, 128, 512);
  k_bnapply<<<gr(4L * 128 * 512), 256, 0, stream>>>(B1, B1, ST, blk[0].g2, blk[0].be2, 128, 512, 4L * 128 * 512, 1);

  // ---- Block 2: C=128, S=8 -> 64@16^3 ----
  k_tohalf_nhwc<<<gr(4L * 128 * 512), 256, 0, stream>>>(B1, AH, 128, 512);
  k_repackW<<<gr(128L * 3456), 256, 0, stream>>>(blk[1].w1, WH, 128, 27, 3456, 128L * 3456);
  k_conv_wmma<128, 128, 8, 3, 1, false, 4><<<dim3(32, 2, 4), 32, 0, stream>>>(AH, WH, blk[1].b1, B0);
  k_bnstats<<<128, 256, 0, stream>>>(B0, ST, 128, 512);
  k_bnapply<<<gr(4L * 128 * 512), 256, 0, stream>>>(B0, B0, ST, blk[1].g1, blk[1].be1, 128, 512, 4L * 128 * 512, 1);
  k_upsample2x<<<gr(4L * 128 * 4096), 256, 0, stream>>>(B0, B1, 128, 8);
  k_tohalf_nhwc<<<gr(4L * 128 * 4096), 256, 0, stream>>>(B1, AH, 128, 4096);
  k_repackW<<<gr(64L * 3456), 256, 0, stream>>>(blk[1].w2, WH, 128, 27, 3456, 64L * 3456);
  k_conv_wmma<128, 64, 16, 3, 1, true, 4><<<dim3(256, 1, 4), 32, 0, stream>>>(AH, WH, blk[1].b2, B0);
  k_bnstats<<<64, 256, 0, stream>>>(B0, ST, 64, 4096);
  k_bnapply<<<gr(4L * 64 * 4096), 256, 0, stream>>>(B0, B0, ST, blk[1].g2, blk[1].be2, 64, 4096, 4L * 64 * 4096, 1);

  // ---- Block 3: C=64, S=16 -> 32@32^3 ----
  k_tohalf_nhwc<<<gr(4L * 64 * 4096), 256, 0, stream>>>(B0, AH, 64, 4096);
  k_repackW<<<gr(64L * 1728), 256, 0, stream>>>(blk[2].w1, WH, 64, 27, 1728, 64L * 1728);
  k_conv_wmma<64, 64, 16, 3, 1, false, 4><<<dim3(256, 1, 4), 32, 0, stream>>>(AH, WH, blk[2].b1, B1);
  k_bnstats<<<64, 256, 0, stream>>>(B1, ST, 64, 4096);
  k_bnapply<<<gr(4L * 64 * 4096), 256, 0, stream>>>(B1, B1, ST, blk[2].g1, blk[2].be1, 64, 4096, 4L * 64 * 4096, 1);
  k_upsample2x<<<gr(4L * 64 * 32768), 256, 0, stream>>>(B1, B0, 64, 16);
  k_tohalf_nhwc<<<gr(4L * 64 * 32768), 256, 0, stream>>>(B0, AH, 64, 32768);
  k_repackW<<<gr(32L * 1728), 256, 0, stream>>>(blk[2].w2, WH, 64, 27, 1728, 32L * 1728);
  k_conv_wmma<64, 32, 32, 3, 1, true, 2><<<dim3(2048, 1, 4), 32, 0, stream>>>(AH, WH, blk[2].b2, B1);
  k_bnstats<<<32, 256, 0, stream>>>(B1, ST, 32, 32768);
  k_bnapply<<<gr(4L * 32 * 32768), 256, 0, stream>>>(B1, B1, ST, blk[2].g2, blk[2].be2, 32, 32768, 4L * 32 * 32768, 1);

  // ---- Block 4: C=32, S=32 -> 16@64^3 ----
  k_tohalf_nhwc<<<gr(4L * 32 * 32768), 256, 0, stream>>>(B1, AH, 32, 32768);
  k_repackW<<<gr(32L * 864), 256, 0, stream>>>(blk[3].w1, WH, 32, 27, 864, 32L * 864);
  k_conv_wmma<32, 32, 32, 3, 1, false, 2><<<dim3(2048, 1, 4), 32, 0, stream>>>(AH, WH, blk[3].b1, B0);
  k_bnstats<<<32, 256, 0, stream>>>(B0, ST, 32, 32768);
  k_bnapply<<<gr(4L * 32 * 32768), 256, 0, stream>>>(B0, B0, ST, blk[3].g1, blk[3].be1, 32, 32768, 4L * 32 * 32768, 1);
  k_upsample2x<<<gr(4L * 32 * 262144), 256, 0, stream>>>(B0, B1, 32, 32);
  k_tohalf_nhwc<<<gr(4L * 32 * 262144), 256, 0, stream>>>(B1, AH, 32, 262144);
  k_repackW<<<gr(16L * 864), 256, 0, stream>>>(blk[3].w2, WH, 32, 27, 864, 16L * 864);
  k_conv_wmma<32, 16, 64, 3, 1, true, 1><<<dim3(16384, 1, 4), 32, 0, stream>>>(AH, WH, blk[3].b2, B0);
  k_bnstats<<<16, 256, 0, stream>>>(B0, ST, 16, 262144);
  k_bnapply<<<gr(4L * 16 * 262144), 256, 0, stream>>>(B0, B0, ST, blk[3].g2, blk[3].be2, 16, 262144, 4L * 16 * 262144, 1);

  // ---- Residual block: two LDS-staged 5^3 convs, C=16, S=64 ----
  k_tohalf_nhwc<<<gr(4L * 16 * 262144), 256, 0, stream>>>(B0, AH, 16, 262144);
  k_repackW<<<gr(16L * 2016), 256, 0, stream>>>(rwa, WH, 16, 125, 2016, 16L * 2016);
  k_conv5_res<<<dim3(4096, 1, 4), 128, 0, stream>>>(AH, WH, rba, B1);
  k_bnstats<<<16, 256, 0, stream>>>(B1, ST, 16, 262144);
  k_bnapply<<<gr(4L * 16 * 262144), 256, 0, stream>>>(B1, B1, ST, rga, rbea, 16, 262144, 4L * 16 * 262144, 1);
  k_tohalf_nhwc<<<gr(4L * 16 * 262144), 256, 0, stream>>>(B1, AH, 16, 262144);
  k_repackW<<<gr(16L * 2016), 256, 0, stream>>>(rwb, WH, 16, 125, 2016, 16L * 2016);
  k_conv5_res<<<dim3(4096, 1, 4), 128, 0, stream>>>(AH, WH, rbb, MD);
  k_bnstats<<<16, 256, 0, stream>>>(MD, ST, 16, 262144);
  k_bnapply<<<gr(4L * 16 * 262144), 256, 0, stream>>>(MD, MD, ST, rgb, rbeb, 16, 262144, 4L * 16 * 262144, 0);
  k_addlrelu<<<gr(4L * 16 * 262144), 256, 0, stream>>>(B0, MD, 4L * 16 * 262144);

  // ---- Head + rotation + compositing ----
  k_outconv_sig<<<gr(4L * 4 * 262144), 256, 0, stream>>>(B0, out_w, out_b, B1);
  k_rotate<<<gr(4L * 262144), 256, 0, stream>>>(B1, angles, MD);
  k_composite<<<gr(4L * 64 * 64), 256, 0, stream>>>(MD, out);
}
